// MoEMessageRouter_76836964926235
// MI455X (gfx1250) — matrix operand order repaired and checked by reading, not compiled, not run on hardware
//
#include <hip/hip_runtime.h>
#include <math.h>

#define E_EDGES   200000
#define DIM       256
#define HID       512
#define NE        8
#define TILE_M    64                              // 4 m-subtiles of 16
#define MI_T      (TILE_M/16)
#define MAX_PAIRS (2*E_EDGES)                     // 400000
#define GATHER_CAP (MAX_PAIRS + NE*TILE_M)        // 400512
#define MAX_TILES (MAX_PAIRS/TILE_M + NE)         // 6258
#define W1_FRAG_HALVES (NE*32*8*512)              // 1048576
#define W2_FRAG_HALVES (NE*16*16*512)             // 1048576

typedef __attribute__((ext_vector_type(16))) _Float16 v16h;
typedef __attribute__((ext_vector_type(8)))  float    v8f;

union Frag16 { v16h v; uint4 q[2]; _Float16 h[16]; };
union F8     { v8f  v; float f[8]; };

// meta layout (ints): [0..7]=counts, [8..15]=fill, [16..24]=segStart, [25..33]=tilePrefix
// ---------------------------------------------------------------------------
__global__ void k_init_out(float* __restrict__ out, int* __restrict__ meta) {
    long long i = (long long)blockIdx.x * blockDim.x + threadIdx.x;
    const long long n4 = (long long)E_EDGES * DIM / 4;
    if (i < n4) ((float4*)out)[i] = make_float4(0.f, 0.f, 0.f, 0.f);
    if (i < 64) meta[i] = 0;
}

__global__ void k_init_gather(int* __restrict__ gTok, float* __restrict__ gProb) {
    int i = blockIdx.x * blockDim.x + threadIdx.x;
    if (i < GATHER_CAP) { gTok[i] = -1; gProb[i] = 0.f; }
}

// Convert fp32 weights to f16, pre-swizzled into WMMA B-fragment layout:
// frag[(e,nt,kc)][lane][h] with B[k][n], n = nt*16 + (lane&15),
// k = kc*32 + (lane<16 ? 0 : 16) + h   (contiguous k order, same as the A loads)
__global__ void k_wprep(const float* __restrict__ w1, const float* __restrict__ w2,
                        _Float16* __restrict__ fragW1, _Float16* __restrict__ fragW2) {
    int i = blockIdx.x * blockDim.x + threadIdx.x;   // 0 .. 2*2^20-1
    int second = (i >= W1_FRAG_HALVES);
    int j = i & (W1_FRAG_HALVES - 1);
    int h    = j & 15;
    int lane = (j >> 4) & 31;
    int nloc = lane & 15;
    int khalf = (lane < 16) ? 0 : 16;
    if (!second) {
        int kc = (j >> 9) & 7;
        int nt = (j >> 12) & 31;
        int e  = j >> 17;
        int n = nt * 16 + nloc;
        int k = kc * 32 + khalf + h;
        fragW1[j] = (_Float16)w1[((long long)e * DIM + k) * HID + n];
    } else {
        int kc = (j >> 9) & 15;
        int nt = (j >> 13) & 15;
        int e  = j >> 17;
        int n = nt * 16 + nloc;
        int k = kc * 32 + khalf + h;
        fragW2[j] = (_Float16)w2[((long long)e * HID + k) * DIM + n];
    }
}

// Gating: one wave32 per token. Lane-parallel dot over DIM, xor-shuffle reduce,
// top-2 + softmax over the two logits, count tokens per expert.
__global__ void k_gate(const float* __restrict__ msg, const float* __restrict__ gw,
                       const float* __restrict__ gb, int* __restrict__ topkIdx,
                       float* __restrict__ topkProb, int* __restrict__ meta) {
    __shared__ float sgw[NE * DIM];
    int tid = threadIdx.x;
    for (int i = tid; i < NE * DIM; i += blockDim.x) sgw[i] = gw[i];
    __syncthreads();
    int wave = tid >> 5, lane = tid & 31;
    int token = blockIdx.x * 8 + wave;
    if (token >= E_EDGES) return;
    float acc[NE];
#pragma unroll
    for (int e = 0; e < NE; e++) acc[e] = 0.f;
    const float* mrow = msg + (long long)token * DIM;
#pragma unroll
    for (int i = 0; i < DIM / 32; i++) {
        float m = mrow[i * 32 + lane];
#pragma unroll
        for (int e = 0; e < NE; e++) acc[e] += m * sgw[e * DIM + i * 32 + lane];
    }
#pragma unroll
    for (int e = 0; e < NE; e++) {
        float v = acc[e];
        for (int off = 16; off >= 1; off >>= 1) v += __shfl_xor(v, off, 32);
        acc[e] = v + gb[e];
    }
    if (lane == 0) {
        int i0 = 0; float v0 = acc[0];
#pragma unroll
        for (int e = 1; e < NE; e++) if (acc[e] > v0) { v0 = acc[e]; i0 = e; }
        int i1 = (i0 == 0) ? 1 : 0; float v1 = acc[i1];
#pragma unroll
        for (int e = 0; e < NE; e++) if (e != i0 && acc[e] > v1) { v1 = acc[e]; i1 = e; }
        float p0 = 1.f / (1.f + expf(v1 - v0));   // softmax over sorted-desc top-2
        float p1 = 1.f - p0;
        topkIdx[token * 2 + 0] = i0; topkIdx[token * 2 + 1] = i1;
        topkProb[token * 2 + 0] = p0; topkProb[token * 2 + 1] = p1;
        atomicAdd(&meta[i0], 1);
        atomicAdd(&meta[i1], 1);
    }
}

// Exclusive scan over padded (multiple-of-TILE_M) per-expert counts.
__global__ void k_scan(int* __restrict__ meta) {
    if (threadIdx.x == 0 && blockIdx.x == 0) {
        int base = 0, tiles = 0;
        for (int e = 0; e < NE; e++) {
            meta[16 + e] = base;
            meta[25 + e] = tiles;
            int c = meta[e];
            int pad = (c + (TILE_M - 1)) & ~(TILE_M - 1);
            base += pad;
            tiles += pad / TILE_M;
            meta[8 + e] = 0;
        }
        meta[16 + NE] = base;
        meta[25 + NE] = tiles;
    }
}

__global__ void k_scatter(const int* __restrict__ topkIdx, const float* __restrict__ topkProb,
                          int* __restrict__ meta, int* __restrict__ gTok,
                          float* __restrict__ gProb) {
    int t = blockIdx.x * blockDim.x + threadIdx.x;
    if (t >= E_EDGES) return;
#pragma unroll
    for (int s = 0; s < 2; s++) {
        int e = topkIdx[t * 2 + s];
        float p = topkProb[t * 2 + s];
        int pos = meta[16 + e] + atomicAdd(&meta[8 + e], 1);
        gTok[pos] = t; gProb[pos] = p;
    }
}

// Main expert-FFN kernel: one 64-token tile per block (4 m-subtiles), 8 waves.
// f16 WMMA with f32 accumulation; ~99 KB LDS (CDNA5 allows 320 KB/workgroup).
__global__ __launch_bounds__(256) void k_moe(const float* __restrict__ msg,
        const float* __restrict__ b1, const float* __restrict__ b2,
        const _Float16* __restrict__ fragW1, const _Float16* __restrict__ fragW2,
        const int* __restrict__ meta, const int* __restrict__ gTok,
        const float* __restrict__ gProb, float* __restrict__ out) {
    __shared__ _Float16 As[TILE_M][DIM + 8];   // 33792 B
    __shared__ _Float16 Hs[TILE_M][HID + 8];   // 66560 B
    __shared__ int   sTok[TILE_M];
    __shared__ float sProb[TILE_M];
    __shared__ int sTP[NE + 1];
    __shared__ int sSS[NE + 1];
    int tid = threadIdx.x;
    if (tid < NE + 1) { sTP[tid] = meta[25 + tid]; sSS[tid] = meta[16 + tid]; }
    __syncthreads();
    int t = blockIdx.x;
    if (t >= sTP[NE]) return;            // uniform across block
    int e = 0;
#pragma unroll
    for (int k = 1; k < NE; k++) if (t >= sTP[k]) e = k;
    int row0 = sSS[e] + (t - sTP[e]) * TILE_M;
    if (tid < TILE_M) { sTok[tid] = gTok[row0 + tid]; sProb[tid] = gProb[row0 + tid]; }
    __syncthreads();

    // Load msg tile (fp32 -> f16) into LDS; padded rows (tok<0) are zero.
    for (int idx = tid; idx < TILE_M * (DIM / 4); idx += 256) {
        int r = idx >> 6;        // DIM/4 == 64
        int c4 = idx & 63;
        int tk = sTok[r];
        float4 v = make_float4(0.f, 0.f, 0.f, 0.f);
        if (tk >= 0) v = ((const float4*)(msg + (long long)tk * DIM))[c4];
        _Float16* dst = &As[r][c4 * 4];
        dst[0] = (_Float16)v.x; dst[1] = (_Float16)v.y;
        dst[2] = (_Float16)v.z; dst[3] = (_Float16)v.w;
    }
    __syncthreads();

    int wave = tid >> 5, lane = tid & 31;
    int m = lane & 15;
    int koff = (lane < 16) ? 0 : 16;    // contiguous K: lane half h holds k = 16h..16h+15
    int rbase = (lane < 16) ? 0 : 8;    // ISA 32-bit C/D layout row split

    // Stage 1: H = gelu(A @ W1 + b1). Each wave owns 4 N-tiles; each B fragment
    // is loaded once from L2 and reused across the 4 m-subtiles.
    const _Float16* w1base = fragW1 + (long long)e * 32 * 8 * 512;
#pragma unroll
    for (int i = 0; i < 4; i++) {
        int nt = wave * 4 + i;
        float b1v = b1[e * HID + nt * 16 + m];
        F8 acc[MI_T];
#pragma unroll
        for (int mi = 0; mi < MI_T; mi++)
#pragma unroll
            for (int r = 0; r < 8; r++) acc[mi].f[r] = b1v;
#pragma unroll
        for (int kc = 0; kc < 8; kc++) {
            const uint4* bw = (const uint4*)(w1base + ((nt * 8 + kc) * 32 + lane) * 16);
            Frag16 bf; bf.q[0] = bw[0]; bf.q[1] = bw[1];
#pragma unroll
            for (int mi = 0; mi < MI_T; mi++) {
                const _Float16* rp = &As[mi * 16 + m][kc * 32 + koff];
                Frag16 af; af.q[0] = *(const uint4*)rp; af.q[1] = *(const uint4*)(rp + 8);
                acc[mi].v = __builtin_amdgcn_wmma_f32_16x16x32_f16(
                    false, af.v, false, bf.v, (short)0, acc[mi].v, false, false);
            }
        }
        int col = nt * 16 + m;
#pragma unroll
        for (int mi = 0; mi < MI_T; mi++)
#pragma unroll
            for (int r = 0; r < 8; r++) {
                float x = acc[mi].f[r];
                float g = 0.5f * x * (1.0f + erff(x * 0.70710678118654752f)); // exact GELU
                Hs[mi * 16 + rbase + r][col] = (_Float16)g;
            }
    }
    __syncthreads();

    // Stage 2: out_tile = (H @ W2 + b2) * prob. Each wave owns 2 N-tiles.
    const _Float16* w2base = fragW2 + (long long)e * 16 * 16 * 512;
#pragma unroll
    for (int i = 0; i < 2; i++) {
        int nt = wave * 2 + i;
        float b2v = b2[e * DIM + nt * 16 + m];
        F8 acc[MI_T];
#pragma unroll
        for (int mi = 0; mi < MI_T; mi++)
#pragma unroll
            for (int r = 0; r < 8; r++) acc[mi].f[r] = b2v;
#pragma unroll
        for (int kc = 0; kc < 16; kc++) {
            const uint4* bw = (const uint4*)(w2base + ((nt * 16 + kc) * 32 + lane) * 16);
            Frag16 bf; bf.q[0] = bw[0]; bf.q[1] = bw[1];
#pragma unroll
            for (int mi = 0; mi < MI_T; mi++) {
                const _Float16* rp = &Hs[mi * 16 + m][kc * 32 + koff];
                Frag16 af; af.q[0] = *(const uint4*)rp; af.q[1] = *(const uint4*)(rp + 8);
                acc[mi].v = __builtin_amdgcn_wmma_f32_16x16x32_f16(
                    false, af.v, false, bf.v, (short)0, acc[mi].v, false, false);
            }
        }
        int col = nt * 16 + m;
#pragma unroll
        for (int mi = 0; mi < MI_T; mi++)
#pragma unroll
            for (int r = 0; r < 8; r++) {
                int row = mi * 16 + rbase + r;
                int tk = sTok[row];
                if (tk >= 0) {
                    float p = sProb[row];
                    atomicAdd(&out[(long long)tk * DIM + col], acc[mi].f[r] * p);
                }
            }
    }
}

// ---------------------------------------------------------------------------
extern "C" void kernel_launch(void* const* d_in, const int* in_sizes, int n_in,
                              void* d_out, int out_size, void* d_ws, size_t ws_size,
                              hipStream_t stream) {
    const float* msg = (const float*)d_in[0];
    const float* gw  = (const float*)d_in[1];
    const float* gb  = (const float*)d_in[2];
    const float* w1  = (const float*)d_in[3];
    const float* b1  = (const float*)d_in[4];
    const float* w2  = (const float*)d_in[5];
    const float* b2  = (const float*)d_in[6];
    float* out = (float*)d_out;

    char* ws = (char*)d_ws;
    size_t off = 0;
    _Float16* fragW1 = (_Float16*)(ws + off); off += (size_t)W1_FRAG_HALVES * 2; // 2 MB
    _Float16* fragW2 = (_Float16*)(ws + off); off += (size_t)W2_FRAG_HALVES * 2; // 2 MB
    int*   topkIdx  = (int*)  (ws + off); off += (size_t)MAX_PAIRS * 4;
    float* topkProb = (float*)(ws + off); off += (size_t)MAX_PAIRS * 4;
    int*   gTok     = (int*)  (ws + off); off += (size_t)GATHER_CAP * 4;
    float* gProb    = (float*)(ws + off); off += (size_t)GATHER_CAP * 4;
    int*   meta     = (int*)  (ws + off); off += 256;

    hipLaunchKernelGGL(k_init_out, dim3(50000), dim3(256), 0, stream, out, meta);
    hipLaunchKernelGGL(k_init_gather, dim3((GATHER_CAP + 255) / 256), dim3(256), 0, stream,
                       gTok, gProb);
    hipLaunchKernelGGL(k_wprep, dim3(8192), dim3(256), 0, stream, w1, w2, fragW1, fragW2);
    hipLaunchKernelGGL(k_gate, dim3(E_EDGES / 8), dim3(256), 0, stream,
                       msg, gw, gb, topkIdx, topkProb, meta);
    hipLaunchKernelGGL(k_scan, dim3(1), dim3(1), 0, stream, meta);
    hipLaunchKernelGGL(k_scatter, dim3((E_EDGES + 255) / 256), dim3(256), 0, stream,
                       topkIdx, topkProb, meta, gTok, gProb);
    hipLaunchKernelGGL(k_moe, dim3(MAX_TILES), dim3(256), 0, stream,
                       msg, b1, b2, fragW1, fragW2, meta, gTok, gProb, out);
}
